// Transformer_hidden_state_learner_69887707840798
// MI455X (gfx1250) — matrix operand order repaired and checked by reading, not compiled, not run on hardware
//
#include <hip/hip_runtime.h>
#include <hip/hip_bf16.h>
#include <stdint.h>

typedef __bf16 bf16_t;
typedef __attribute__((ext_vector_type(16))) __bf16 v16bf;
typedef __attribute__((ext_vector_type(8)))  float  v8f;
typedef int v4i __attribute__((vector_size(16)));   // GCC-style, matches builtin proto

#if defined(__HIP_DEVICE_COMPILE__) && __has_builtin(__builtin_amdgcn_global_load_async_to_lds_b128) && __has_builtin(__builtin_amdgcn_s_wait_asynccnt)
#define USE_ASYNC_LDS 1
#else
#define USE_ASYNC_LDS 0
#endif

namespace {
constexpr int    kD     = 1024;
constexpr int    kBatch = 16;
constexpr int    kSeq   = 512;
constexpr int    kRows  = kBatch * kSeq;   // 8192
constexpr int    kKT    = kD / 32;         // 32 K-steps of depth 32
constexpr float  kNegInf = -1.0e9f;
constexpr float  kScale  = 0.125f;         // 1/sqrt(64)
constexpr int    kBP    = 40;              // LDS B-tile pitch in bf16 (80 B)

// workspace layout (bytes)
constexpr size_t OFF_XBF = 0;                   // [8192,1024] bf16 activations
constexpr size_t OFF_WTQ = 16777216;            // [1024,1024] bf16, transposed [out][in]
constexpr size_t OFF_WTK = OFF_WTQ + 2097152;
constexpr size_t OFF_WTV = OFF_WTK + 2097152;
constexpr size_t OFF_WTO = OFF_WTV + 2097152;
constexpr size_t OFF_QH  = OFF_WTO + 2097152;   // [B,H,S,Dh] bf16
constexpr size_t OFF_KH  = OFF_QH  + 16777216;  // [B,H,S,Dh] bf16
constexpr size_t OFF_VT  = OFF_KH  + 16777216;  // [B,H,Dh,S] bf16 (transposed)
constexpr size_t OFF_OBF = OFF_VT  + 16777216;  // [8192,1024] bf16 attention output
} // namespace

union FragU { uint4 u[2]; v16bf v; };

__device__ __forceinline__ v8f wmma_bf16(v16bf a, v16bf b, v8f c) {
  return __builtin_amdgcn_wmma_f32_16x16x32_bf16(false, a, false, b, (short)0, c,
                                                 false, false);
}

// A fragment (16x32 bf16). base -> (row0, k0) of row-major [*, ld] bf16.
__device__ __forceinline__ v16bf load_a(const bf16_t* base, int ld, int lane) {
  const int m = lane & 15, half = lane >> 4;
  const char* p = (const char*)(base + (size_t)m * ld);
  FragU f;
  f.u[0] = *(const uint4*)(p + 32 * half);
  f.u[1] = *(const uint4*)(p + 32 * half + 16);
  return f.v;
}

// B fragment (32x16 bf16) from a "columns as rows" matrix with pitch ld elems.
__device__ __forceinline__ v16bf load_b(const bf16_t* base, int ld, int lane) {
  const int n = lane & 15, half = lane >> 4;
  const char* p = (const char*)(base + (size_t)n * ld) + 32 * half;
  FragU f;
  f.u[0] = *(const uint4*)(p);
  f.u[1] = *(const uint4*)(p + 16);
  return f.v;
}

// ---- async LDS staging helpers (GLOBAL_LOAD_ASYNC_TO_LDS_B128 / ASYNCcnt) ----
__device__ __forceinline__ void lds_fill16(const bf16_t* gsrc, bf16_t* ldst) {
#if USE_ASYNC_LDS
  typedef __attribute__((address_space(1))) v4i* g_v4i_p;
  typedef __attribute__((address_space(3))) v4i* l_v4i_p;
  __builtin_amdgcn_global_load_async_to_lds_b128((g_v4i_p)gsrc, (l_v4i_p)ldst, 0, 0);
#else
  *(uint4*)ldst = *(const uint4*)gsrc;
#endif
}
__device__ __forceinline__ void wait_async1() {
#if USE_ASYNC_LDS
  __builtin_amdgcn_s_wait_asynccnt(1);
#endif
}
__device__ __forceinline__ void wait_async0() {
#if USE_ASYNC_LDS
  __builtin_amdgcn_s_wait_asynccnt(0);
#endif
}

// B fragment from the padded LDS tile (pitch kBP elems = 80 B).
__device__ __forceinline__ v16bf load_b_lds(const bf16_t* base, int lane) {
  const int n = lane & 15, half = lane >> 4;
  const char* p = (const char*)(base + n * kBP) + 32 * half;
  FragU f;
  f.u[0] = *(const uint4*)(p);
  f.u[1] = *(const uint4*)(p + 16);
  return f.v;
}

// ---------------- staging kernels ----------------
__global__ void __launch_bounds__(256)
k_cvt(const float* __restrict__ src, bf16_t* __restrict__ dst, int n) {
  int i = blockIdx.x * blockDim.x + threadIdx.x;
  const int stride = gridDim.x * blockDim.x;
  for (; i < n; i += stride) dst[i] = (bf16_t)src[i];
}

__global__ void __launch_bounds__(256)
k_cvtT(const float* __restrict__ W, bf16_t* __restrict__ Wt) {
  const int i = blockIdx.x * blockDim.x + threadIdx.x;
  if (i >= kD * kD) return;
  const int k = i >> 10, n = i & 1023;                    // W[k][n] row-major
  Wt[(size_t)n * kD + k] = (bf16_t)W[i];
}

// ---------------- fused QKV projection ----------------
// Block: 8 waves = 128 rows x 64 cols. B tile (64x32) staged in double-buffered
// LDS via async copies and shared by all waves; A fragments register-pipelined.
__global__ void __launch_bounds__(256)
k_qkv(const bf16_t* __restrict__ X,
      const bf16_t* __restrict__ WtQ, const bf16_t* __restrict__ WtK,
      const bf16_t* __restrict__ WtV,
      const float* __restrict__ bq, const float* __restrict__ bk,
      const float* __restrict__ bv,
      bf16_t* __restrict__ Qh, bf16_t* __restrict__ Kh, bf16_t* __restrict__ Vt) {
  __shared__ __align__(16) bf16_t Btile[2][64 * kBP];
  const int tid = threadIdx.x;
  const int lane = tid & 31, wv = tid >> 5;
  const int bc = blockIdx.x;
  const int ct = bc & 15;            // 16 col-tiles of 64
  const int rb = (bc >> 4) & 63;     // 64 row-blocks of 128
  const int z  = bc >> 10;           // 0=Q 1=K 2=V
  const int c0 = ct * 64;
  const int r0 = rb * 128 + wv * 16;
  const bf16_t* Wt  = (z == 0) ? WtQ : (z == 1) ? WtK : WtV;
  const float* bias = (z == 0) ? bq  : (z == 1) ? bk  : bv;

  // fill mapping: 256 threads move 64 rows x 32 bf16 (one b128 each)
  const int frow = tid >> 2, fch = tid & 3;
  const bf16_t* fsrc = Wt + (size_t)(c0 + frow) * kD + fch * 8;
  bf16_t* fd0 = &Btile[0][frow * kBP + fch * 8];
  bf16_t* fd1 = &Btile[1][frow * kBP + fch * 8];

  lds_fill16(fsrc, fd0);                       // stage K-step 0
  const bf16_t* xrow = X + (size_t)r0 * kD;
  v16bf a_cur = load_a(xrow, kD, lane);
  v8f acc[4] = { {}, {}, {}, {} };

  for (int t = 0; t < kKT; ++t) {
    const int kk = t * 32;
    if (t + 1 < kKT) {                         // prefetch next B tile
      lds_fill16(fsrc + kk + 32, (t & 1) ? fd0 : fd1);
      wait_async1();                           // only older transfer must land
    } else {
      wait_async0();
    }
    __syncthreads();
    v16bf a_nxt = a_cur;
    if (t + 1 < kKT) a_nxt = load_a(xrow + kk + 32, kD, lane);
    const bf16_t* bb = Btile[t & 1];
#pragma unroll
    for (int j = 0; j < 4; ++j) {
      const v16bf b = load_b_lds(bb + (16 * j) * kBP, lane);
      acc[j] = wmma_bf16(a_cur, b, acc[j]);
    }
    a_cur = a_nxt;
    __syncthreads();
  }

  const int n = lane & 15, half = lane >> 4;
#pragma unroll
  for (int j = 0; j < 4; ++j) {
    const int col = c0 + 16 * j + n;
    const float bb = bias[col];
    const int hh = col >> 6, dh = col & 63;
#pragma unroll
    for (int g = 0; g < 8; ++g) {
      const int r = r0 + g + 8 * half;
      const int bi = r >> 9, s = r & 511;
      const bf16_t val = (bf16_t)(acc[j][g] + bb);
      if (z == 0)      Qh[(((size_t)bi * 16 + hh) * 512 + s) * 64 + dh] = val;
      else if (z == 1) Kh[(((size_t)bi * 16 + hh) * 512 + s) * 64 + dh] = val;
      else             Vt[(((size_t)bi * 16 + hh) * 64 + dh) * 512 + s] = val;
    }
  }
}

// ---------------- flash attention, one wave per (b,h,16-query tile) ----------------
__global__ void __launch_bounds__(128)
k_attn(const bf16_t* __restrict__ Qh, const bf16_t* __restrict__ Kh,
       const bf16_t* __restrict__ Vt, const int* __restrict__ lengths,
       bf16_t* __restrict__ Obf) {
  __shared__ __align__(16) bf16_t pT[4][16 * 40];   // 80B pitch/row, per wave
  const int lane = threadIdx.x & 31;
  const int wv   = threadIdx.x >> 5;
  const int gw   = blockIdx.x * 4 + wv;
  const int qt = gw & 31, h = (gw >> 5) & 15, b = gw >> 9;
  const int q0 = qt * 16;
  const size_t bh = (size_t)b * 16 + h;
  const bf16_t* Qb = Qh + bh * (512 * 64);
  const bf16_t* Kb = Kh + bh * (512 * 64);
  const bf16_t* Vb = Vt + bh * (64 * 512);
  bf16_t* myP = pT[wv];

  const int n = lane & 15, half = lane >> 4;
  const int len = lengths[b];
  const int ntiles = (len + 31) >> 5;               // >=1 since len>=1

  const v16bf aq0 = load_a(Qb + (size_t)q0 * 64, 64, lane);       // dh 0..31
  const v16bf aq1 = load_a(Qb + (size_t)q0 * 64 + 32, 64, lane);  // dh 32..63

  v8f acc[4] = { {}, {}, {}, {} };
  float mrow[8], lrow[8];
#pragma unroll
  for (int g = 0; g < 8; ++g) { mrow[g] = -3.0e38f; lrow[g] = 0.0f; }

  for (int kt = 0; kt < ntiles; ++kt) {
    const int k0 = kt * 32;
    if (kt + 1 < ntiles) {                          // global_prefetch_b8
      __builtin_prefetch(Kb + (size_t)(k0 + 32 + n) * 64 + 32 * half, 0, 1);
      __builtin_prefetch(Vb + (size_t)n * 512 + k0 + 32 + 16 * half, 0, 1);
    }
    v8f s0 = {}, s1 = {};
    s0 = wmma_bf16(aq0, load_b(Kb + (size_t)k0 * 64, 64, lane), s0);
    s0 = wmma_bf16(aq1, load_b(Kb + (size_t)k0 * 64 + 32, 64, lane), s0);
    s1 = wmma_bf16(aq0, load_b(Kb + (size_t)(k0 + 16) * 64, 64, lane), s1);
    s1 = wmma_bf16(aq1, load_b(Kb + (size_t)(k0 + 16) * 64 + 32, 64, lane), s1);

    const bool val0 = (k0 + n) < len;
    const bool val1 = (k0 + 16 + n) < len;

#pragma unroll
    for (int g = 0; g < 8; ++g) {
      const float e0 = val0 ? s0[g] * kScale : kNegInf;
      const float e1 = val1 ? s1[g] * kScale : kNegInf;
      float t = fmaxf(e0, e1);                       // row max over 32 keys
      t = fmaxf(t, __shfl_xor(t, 1));
      t = fmaxf(t, __shfl_xor(t, 2));
      t = fmaxf(t, __shfl_xor(t, 4));
      t = fmaxf(t, __shfl_xor(t, 8));
      const float mnew = fmaxf(mrow[g], t);
      const float corr = __expf(mrow[g] - mnew);
      const float p0 = __expf(e0 - mnew);
      const float p1 = __expf(e1 - mnew);
      float rs = p0 + p1;                            // row sum
      rs += __shfl_xor(rs, 1);
      rs += __shfl_xor(rs, 2);
      rs += __shfl_xor(rs, 4);
      rs += __shfl_xor(rs, 8);
      lrow[g] = lrow[g] * corr + rs;
      mrow[g] = mnew;
#pragma unroll
      for (int c = 0; c < 4; ++c) acc[c][g] *= corr;
      const int row = g + 8 * half;                  // C-layout -> LDS tile
      myP[row * 40 + n]      = (bf16_t)p0;
      myP[row * 40 + 16 + n] = (bf16_t)p1;
    }

    FragU f;                                         // P tile as A fragment
    {
      const char* p = (const char*)myP + (lane & 15) * 80;
      f.u[0] = *(const uint4*)(p + 32 * half);
      f.u[1] = *(const uint4*)(p + 32 * half + 16);
    }
#pragma unroll
    for (int c = 0; c < 4; ++c) {
      const v16bf bv = load_b(Vb + (size_t)(c * 16) * 512 + k0, 512, lane);
      acc[c] = wmma_bf16(f.v, bv, acc[c]);
    }
  }

  float inv[8];
#pragma unroll
  for (int g = 0; g < 8; ++g) inv[g] = 1.0f / lrow[g];
#pragma unroll
  for (int c = 0; c < 4; ++c) {
    const int col = h * 64 + c * 16 + n;
#pragma unroll
    for (int g = 0; g < 8; ++g) {
      const int s = q0 + g + 8 * half;
      Obf[((size_t)b * 512 + s) * kD + col] = (bf16_t)(acc[c][g] * inv[g]);
    }
  }
}

// ---------------- output projection (same block-cooperative scheme) ----------------
__global__ void __launch_bounds__(256)
k_ogemm(const bf16_t* __restrict__ Ob, const bf16_t* __restrict__ WtO,
        const float* __restrict__ bo, float* __restrict__ out) {
  __shared__ __align__(16) bf16_t Btile[2][64 * kBP];
  const int tid = threadIdx.x;
  const int lane = tid & 31, wv = tid >> 5;
  const int bc = blockIdx.x;
  const int ct = bc & 15;
  const int rb = bc >> 4;            // 64 row-blocks
  const int c0 = ct * 64;
  const int r0 = rb * 128 + wv * 16;

  const int frow = tid >> 2, fch = tid & 3;
  const bf16_t* fsrc = WtO + (size_t)(c0 + frow) * kD + fch * 8;
  bf16_t* fd0 = &Btile[0][frow * kBP + fch * 8];
  bf16_t* fd1 = &Btile[1][frow * kBP + fch * 8];

  lds_fill16(fsrc, fd0);
  const bf16_t* orow = Ob + (size_t)r0 * kD;
  v16bf a_cur = load_a(orow, kD, lane);
  v8f acc[4] = { {}, {}, {}, {} };

  for (int t = 0; t < kKT; ++t) {
    const int kk = t * 32;
    if (t + 1 < kKT) {
      lds_fill16(fsrc + kk + 32, (t & 1) ? fd0 : fd1);
      wait_async1();
    } else {
      wait_async0();
    }
    __syncthreads();
    v16bf a_nxt = a_cur;
    if (t + 1 < kKT) a_nxt = load_a(orow + kk + 32, kD, lane);
    const bf16_t* bb = Btile[t & 1];
#pragma unroll
    for (int j = 0; j < 4; ++j) {
      const v16bf b = load_b_lds(bb + (16 * j) * kBP, lane);
      acc[j] = wmma_bf16(a_cur, b, acc[j]);
    }
    a_cur = a_nxt;
    __syncthreads();
  }

  const int n = lane & 15, half = lane >> 4;
#pragma unroll
  for (int j = 0; j < 4; ++j) {
    const int col = c0 + 16 * j + n;
    const float bb = bo[col];
#pragma unroll
    for (int g = 0; g < 8; ++g) {
      const int r = r0 + g + 8 * half;
      out[(size_t)r * kD + col] = acc[j][g] + bb;
    }
  }
}

extern "C" void kernel_launch(void* const* d_in, const int* in_sizes, int n_in,
                              void* d_out, int out_size, void* d_ws, size_t ws_size,
                              hipStream_t stream) {
  (void)in_sizes; (void)n_in; (void)out_size; (void)ws_size;
  const float* v       = (const float*)d_in[0];
  const int*   lengths = (const int*)d_in[1];
  const float* Wq = (const float*)d_in[2];
  const float* bq = (const float*)d_in[3];
  const float* Wk = (const float*)d_in[4];
  const float* bk = (const float*)d_in[5];
  const float* Wv = (const float*)d_in[6];
  const float* bv = (const float*)d_in[7];
  const float* Wo = (const float*)d_in[8];
  const float* bo = (const float*)d_in[9];

  char* ws = (char*)d_ws;
  bf16_t* XBF = (bf16_t*)(ws + OFF_XBF);
  bf16_t* WTQ = (bf16_t*)(ws + OFF_WTQ);
  bf16_t* WTK = (bf16_t*)(ws + OFF_WTK);
  bf16_t* WTV = (bf16_t*)(ws + OFF_WTV);
  bf16_t* WTO = (bf16_t*)(ws + OFF_WTO);
  bf16_t* QH  = (bf16_t*)(ws + OFF_QH);
  bf16_t* KH  = (bf16_t*)(ws + OFF_KH);
  bf16_t* VT  = (bf16_t*)(ws + OFF_VT);
  bf16_t* OBF = (bf16_t*)(ws + OFF_OBF);

  k_cvt<<<2048, 256, 0, stream>>>(v, XBF, kRows * kD);
  k_cvtT<<<4096, 256, 0, stream>>>(Wq, WTQ);
  k_cvtT<<<4096, 256, 0, stream>>>(Wk, WTK);
  k_cvtT<<<4096, 256, 0, stream>>>(Wv, WTV);
  k_cvtT<<<4096, 256, 0, stream>>>(Wo, WTO);
  // 3 mats * 64 row-blocks * 16 col-tiles
  k_qkv<<<3072, 256, 0, stream>>>(XBF, WTQ, WTK, WTV, bq, bk, bv, QH, KH, VT);
  // 16*16*32 = 8192 waves / 4 per block
  k_attn<<<2048, 128, 0, stream>>>(QH, KH, VT, lengths, OBF);
  // 64 row-blocks * 16 col-tiles
  k_ogemm<<<1024, 256, 0, stream>>>(OBF, WTO, bo, (float*)d_out);
}
